// LidarYOLOX_58256936403432
// MI455X (gfx1250) — matrix operand order repaired and checked by reading, not compiled, not run on hardware
//
#include <hip/hip_runtime.h>
#include <hip/hip_bf16.h>

typedef __attribute__((ext_vector_type(16))) _Float16 v16h;
typedef __attribute__((ext_vector_type(8)))  float    v8f;

#define N_PTS   65536
#define N_EDGE  262144
#define N_MSG   (N_EDGE + N_PTS)     // 327680, multiple of 64
#define BATCH   4
#define HFT     60
#define WFT     80
#define NPIX    (BATCH*HFT*WFT)      // 19200, multiple of 64
#define SIMG    8

// ---------------------------------------------------------------------------
// CDNA5 async global->LDS copy: 64 contiguous bytes per lane (4 x b128),
// tracked with ASYNCcnt (no VGPR data path, no loadcnt stall).
// ---------------------------------------------------------------------------
__device__ __forceinline__ void async_copy64(unsigned lds_off, const _Float16* g)
{
    asm volatile(
        "global_load_async_to_lds_b128 %0, %1, off\n\t"
        "global_load_async_to_lds_b128 %0, %1, off offset:16\n\t"
        "global_load_async_to_lds_b128 %0, %1, off offset:32\n\t"
        "global_load_async_to_lds_b128 %0, %1, off offset:48"
        :: "v"(lds_off), "v"(g) : "memory");
}
__device__ __forceinline__ void wait_async0()
{
    asm volatile("s_wait_asynccnt 0x0" ::: "memory");
}
__device__ __forceinline__ unsigned lds_off32(const void* p)
{   // flat shared addr = {aperture, lds_offset}; low 32 bits are the LDS offset
    return (unsigned)(size_t)p;
}

// ---------------------------------------------------------------------------
// 64x64-tile WMMA GEMM, K-chunk 64, double-buffered async tile loads.
// Out[M x Nc] = act(A[M x K] * Bt[Nc x K]^T + bias); K multiple of 64.
// 128 threads = 4 waves; wave w -> rows [w*16, w*16+16) x 64 cols.
// ---------------------------------------------------------------------------
template<int ACT>   // 0=none 1=relu 2=silu
__global__ void __launch_bounds__(128)
wmma_gemm(const _Float16* __restrict__ A, int M, int K,
          const _Float16* __restrict__ Bt,
          const float* __restrict__ bias,
          _Float16* __restrict__ Out, int Nc)
{
    __shared__ __align__(16) _Float16 As[2][64][64];
    __shared__ __align__(16) _Float16 Bs[2][64][64];
    const int tid  = threadIdx.x;
    const int wave = tid >> 5;
    const int lane = tid & 31;
    const long tileM = (long)blockIdx.x * 64;
    const int  tileN = blockIdx.y * 64;
    const int row  = tid >> 1;
    const int hsel = tid & 1;

    long mA = tileM + row; if (mA >= M) mA = M - 1;
    const _Float16* gA = A  + mA * (long)K + hsel * 32;
    const _Float16* gB = Bt + (long)(tileN + row) * K + hsel * 32;
    const unsigned ldsA[2] = { lds_off32(&As[0][row][hsel*32]), lds_off32(&As[1][row][hsel*32]) };
    const unsigned ldsB[2] = { lds_off32(&Bs[0][row][hsel*32]), lds_off32(&Bs[1][row][hsel*32]) };

    v8f acc[4] = {};
    const int nk = K >> 6;

    async_copy64(ldsA[0], gA);
    async_copy64(ldsB[0], gB);

    for (int i = 0; i < nk; ++i) {
        wait_async0();          // this wave's tile-i copies done
        __syncthreads();        // everyone's tile i ready; prev compute finished
        if (i + 1 < nk) {
            async_copy64(ldsA[(i+1)&1], gA + (long)(i+1)*64);
            async_copy64(ldsB[(i+1)&1], gB + (long)(i+1)*64);
        }
        const _Float16 (*Ab)[64] = As[i & 1];
        const _Float16 (*Bb)[64] = Bs[i & 1];
#pragma unroll
        for (int sub = 0; sub < 2; ++sub) {
            union { v16h v; unsigned u[8]; } afr;
            const int mr = wave * 16 + (lane & 15);
            const int hg = (lane >> 4) * 16;
#pragma unroll
            for (int p = 0; p < 8; ++p) {
                int k0 = sub*32 + 2*p + hg;
                afr.u[p] = *(const unsigned*)&Ab[mr][k0];
            }
#pragma unroll
            for (int nb = 0; nb < 4; ++nb) {
                union { v16h v; unsigned u[8]; } bfr;
                const int nl = nb * 16 + (lane & 15);
                const int kb = sub*32 + (lane >> 4) * 16;
#pragma unroll
                for (int p = 0; p < 8; ++p)
                    bfr.u[p] = *(const unsigned*)&Bb[nl][kb + 2*p];
                acc[nb] = __builtin_amdgcn_wmma_f32_16x16x32_f16(
                    false, afr.v, false, bfr.v, (short)0, acc[nb], false, false);
            }
        }
    }

#pragma unroll
    for (int nb = 0; nb < 4; ++nb) {
        int n = tileN + nb * 16 + (lane & 15);
        float bv = bias ? bias[n] : 0.f;
#pragma unroll
        for (int r = 0; r < 8; ++r) {
            long m = tileM + wave * 16 + ((lane < 16) ? r : r + 8);
            if (m >= M) continue;
            float f = acc[nb][r] + bv;
            if (ACT == 1) f = fmaxf(f, 0.f);
            else if (ACT == 2) f = f / (1.f + __expf(-f));
            Out[m * (long)Nc + n] = (_Float16)f;
        }
    }
}

// ---------------------------------------------------------------------------
// Implicit-GEMM 3x3 conv (pad=1), NHWC f16 (BATCH,HFT,WFT,256), K = 9*256.
// Weights async->LDS; activations register-staged (halo needs zero-fill).
// Double-buffered, one barrier per 64-K chunk.
// ---------------------------------------------------------------------------
template<int ACT>
__global__ void __launch_bounds__(128)
wmma_conv3(const _Float16* __restrict__ In,
           const _Float16* __restrict__ Bt,   // [co][2304], K=(ky*3+kx)*256+ci
           const float* __restrict__ bias,
           _Float16* __restrict__ Out)
{
    __shared__ __align__(16) _Float16 As[2][64][64];
    __shared__ __align__(16) _Float16 Bs[2][64][64];
    const int tid  = threadIdx.x;
    const int wave = tid >> 5;
    const int lane = tid & 31;
    const int tileM = blockIdx.x * 64;   // pixel tile
    const int tileN = blockIdx.y * 64;   // cout tile
    const int row  = tid >> 1;
    const int hsel = tid & 1;

    const int p  = tileM + row;
    const int b  = p / (HFT * WFT);
    const int rm = p - b * (HFT * WFT);
    const int y  = rm / WFT;
    const int x  = rm - y * WFT;

    const _Float16* gB = Bt + (long)(tileN + row) * 2304 + hsel * 32;
    const unsigned ldsA[2] = { lds_off32(&As[0][row][hsel*32]), lds_off32(&As[1][row][hsel*32]) };
    const unsigned ldsB[2] = { lds_off32(&Bs[0][row][hsel*32]), lds_off32(&Bs[1][row][hsel*32]) };

    auto loadA = [&](int kk, float4* ar) {
        const int tap = kk >> 8;
        const int ci  = (kk & 255) + hsel * 32;
        const int ky = tap / 3, kx = tap - ky * 3;
        const int yy = y + ky - 1, xx = x + kx - 1;
        float4 z = {0.f,0.f,0.f,0.f};
        ar[0] = ar[1] = ar[2] = ar[3] = z;
        if (yy >= 0 && yy < HFT && xx >= 0 && xx < WFT) {
            const float4* src = (const float4*)(In +
                ((((long)b * HFT + yy) * WFT + xx) * 256 + ci));
            ar[0] = src[0]; ar[1] = src[1]; ar[2] = src[2]; ar[3] = src[3];
        }
    };

    v8f acc[4] = {};
    const int nk = 2304 >> 6;     // 36

    float4 ar[4];
    loadA(0, ar);
    { float4* d = (float4*)(&As[0][row][hsel*32]);
      d[0]=ar[0]; d[1]=ar[1]; d[2]=ar[2]; d[3]=ar[3]; }
    async_copy64(ldsB[0], gB);

    for (int i = 0; i < nk; ++i) {
        wait_async0();
        __syncthreads();
        const int nxt = (i + 1) & 1;
        if (i + 1 < nk) {
            async_copy64(ldsB[nxt], gB + (long)(i+1)*64);
            loadA((i+1) * 64, ar);
        }
        const _Float16 (*Ab)[64] = As[i & 1];
        const _Float16 (*Bb)[64] = Bs[i & 1];
#pragma unroll
        for (int sub = 0; sub < 2; ++sub) {
            union { v16h v; unsigned u[8]; } afr;
            const int mr = wave * 16 + (lane & 15);
            const int hg = (lane >> 4) * 16;
#pragma unroll
            for (int q = 0; q < 8; ++q) {
                int k0 = sub*32 + 2*q + hg;
                afr.u[q] = *(const unsigned*)&Ab[mr][k0];
            }
#pragma unroll
            for (int nb = 0; nb < 4; ++nb) {
                union { v16h v; unsigned u[8]; } bfr;
                const int nl = nb * 16 + (lane & 15);
                const int kb = sub*32 + (lane >> 4) * 16;
#pragma unroll
                for (int q = 0; q < 8; ++q)
                    bfr.u[q] = *(const unsigned*)&Bb[nl][kb + 2*q];
                acc[nb] = __builtin_amdgcn_wmma_f32_16x16x32_f16(
                    false, afr.v, false, bfr.v, (short)0, acc[nb], false, false);
            }
        }
        if (i + 1 < nk) {   // stage next activation tile into the other buffer
            float4* d = (float4*)(&As[nxt][row][hsel*32]);
            d[0]=ar[0]; d[1]=ar[1]; d[2]=ar[2]; d[3]=ar[3];
        }
    }

#pragma unroll
    for (int nb = 0; nb < 4; ++nb) {
        int n = tileN + nb * 16 + (lane & 15);
        float bv = bias[n];
#pragma unroll
        for (int r = 0; r < 8; ++r) {
            long m = tileM + wave * 16 + ((lane < 16) ? r : r + 8);
            float f = acc[nb][r] + bv;
            if (ACT == 1) f = fmaxf(f, 0.f);
            else if (ACT == 2) f = f / (1.f + __expf(-f));
            Out[m * 256 + n] = (_Float16)f;
        }
    }
}

// ---------------------------------------------------------------------------
// Support kernels
// ---------------------------------------------------------------------------
__global__ void cvt_lin_w(const float* __restrict__ w, int fan_out, int fan_in,
                          int Kpad, _Float16* __restrict__ Bt)
{
    long idx = (long)blockIdx.x * blockDim.x + threadIdx.x;
    if (idx >= (long)fan_out * Kpad) return;
    int n = idx / Kpad, k = idx % Kpad;
    Bt[idx] = (k < fan_in) ? (_Float16)w[(long)n * fan_in + k] : (_Float16)0.f;
}

__global__ void cvt_conv_w(const float* __restrict__ w, _Float16* __restrict__ Bt)
{   // w: (256,256,3,3) -> Bt[co][(ky*3+kx)*256+ci]
    long idx = (long)blockIdx.x * blockDim.x + threadIdx.x;
    if (idx >= 256L * 2304) return;
    int co = idx / 2304, r = idx % 2304;
    int tap = r >> 8, ci = r & 255;
    int ky = tap / 3, kx = tap - ky * 3;
    Bt[idx] = (_Float16)w[(((long)co * 256 + ci) * 3 + ky) * 3 + kx];
}

__global__ void gather_msgs(const float* __restrict__ x0,
                            const _Float16* __restrict__ hprev,
                            int din, int Kpad,
                            const float* __restrict__ pos,
                            const int* __restrict__ ei,
                            _Float16* __restrict__ A)
{
    long idx = (long)blockIdx.x * blockDim.x + threadIdx.x;
    if (idx >= (long)N_MSG * Kpad) return;
    int m = idx / Kpad, c = idx % Kpad;
    int src = (m < N_EDGE) ? ei[m] : (m - N_EDGE);
    float val = 0.f;
    if (c < din) {
        val = x0 ? x0[(long)src * din + c] : (float)hprev[(long)src * din + c];
    } else if (c < din + 3) {
        int dst = (m < N_EDGE) ? ei[N_EDGE + m] : (m - N_EDGE);
        val = pos[(long)src * 3 + (c - din)] - pos[(long)dst * 3 + (c - din)];
    }
    A[idx] = (_Float16)val;
}

__global__ void fill_u32(unsigned* p, unsigned v, long n)
{ long i = (long)blockIdx.x * blockDim.x + threadIdx.x; if (i < n) p[i] = v; }

__global__ void fill_f32(float* p, float v, long n)
{ long i = (long)blockIdx.x * blockDim.x + threadIdx.x; if (i < n) p[i] = v; }

__global__ void segmax_scatter(const _Float16* __restrict__ G, int H,
                               const int* __restrict__ ei,
                               unsigned* __restrict__ keys)
{
    int m = blockIdx.x, c = threadIdx.x;
    int dst = (m < N_EDGE) ? ei[N_EDGE + m] : (m - N_EDGE);
    float f = (float)G[(long)m * H + c];
    unsigned u = __float_as_uint(f);
    unsigned key = (u & 0x80000000u) ? ~u : (u | 0x80000000u);
    atomicMax(&keys[(long)dst * H + c], key);
}

__global__ void keys_to_f16(const unsigned* __restrict__ keys,
                            _Float16* __restrict__ h, long n)
{
    long i = (long)blockIdx.x * blockDim.x + threadIdx.x;
    if (i >= n) return;
    unsigned key = keys[i];
    unsigned u = (key & 0x80000000u) ? (key & 0x7FFFFFFFu) : ~key;
    h[i] = (_Float16)__uint_as_float(u);
}

__global__ void scatter_image(const _Float16* __restrict__ h,
                              const float* __restrict__ pos,
                              const int* __restrict__ batch,
                              const float* __restrict__ Kmat,
                              float* __restrict__ fm)
{
    int n = blockIdx.x, c = threadIdx.x;
    float X = pos[n * 3 + 0], Y = pos[n * 3 + 1], Z = pos[n * 3 + 2];
    float fx = Kmat[0], cx = Kmat[2], fy = Kmat[4], cy = Kmat[5];
    int u = (int)((fx * X / Z + cx) / (float)SIMG);
    int v = (int)((fy * Y / Z + cy) / (float)SIMG);
    if (u < 0 || u >= WFT || v < 0 || v >= HFT) return;
    long base = (((long)batch[n] * HFT + v) * WFT + u) * 256;
    atomicAdd(&fm[base + c], (float)h[(long)n * 256 + c]);
}

__global__ void f32_to_f16(const float* __restrict__ in, _Float16* __restrict__ out, long n)
{ long i = (long)blockIdx.x * blockDim.x + threadIdx.x; if (i < n) out[i] = (_Float16)in[i]; }

__global__ void pred_decode(const _Float16* __restrict__ cfeat,
                            const _Float16* __restrict__ rfeat,
                            const float* __restrict__ clsp_w, const float* __restrict__ clsp_b,
                            const float* __restrict__ regp_w, const float* __restrict__ regp_b,
                            const float* __restrict__ objp_w, const float* __restrict__ objp_b,
                            float* __restrict__ out)
{
    long idx = (long)blockIdx.x * blockDim.x + threadIdx.x;
    if (idx >= (long)NPIX * 8) return;
    int p = idx >> 3, ch = idx & 7;
    const _Float16* feat; const float* w; float b;
    if (ch < 4)      { feat = rfeat; w = regp_w + ch * 256;       b = regp_b[ch]; }
    else if (ch == 4){ feat = rfeat; w = objp_w;                  b = objp_b[0]; }
    else             { feat = cfeat; w = clsp_w + (ch - 5) * 256; b = clsp_b[ch - 5]; }
    float s = b;
    const _Float16* f = feat + (long)p * 256;
    for (int i = 0; i < 256; ++i) s += (float)f[i] * w[i];
    int pp = p % (HFT * WFT);
    int y = pp / WFT, x = pp - y * WFT;
    float val;
    if (ch == 0)      val = (s + (float)x) * (float)SIMG;
    else if (ch == 1) val = (s + (float)y) * (float)SIMG;
    else if (ch <= 3) val = __expf(s) * (float)SIMG;
    else              val = 1.f / (1.f + __expf(-s));
    out[idx] = val;
}

// ---------------------------------------------------------------------------
extern "C" void kernel_launch(void* const* d_in, const int* in_sizes, int n_in,
                              void* d_out, int out_size, void* d_ws, size_t ws_size,
                              hipStream_t stream)
{
    const float* x     = (const float*)d_in[0];
    const float* pos   = (const float*)d_in[1];
    const int*   ei    = (const int*)d_in[2];
    const int*   batch = (const int*)d_in[3];
    const float* Kmat  = (const float*)d_in[4];
    const float* W1[3] = {(const float*)d_in[5],(const float*)d_in[9],(const float*)d_in[13]};
    const float* B1[3] = {(const float*)d_in[6],(const float*)d_in[10],(const float*)d_in[14]};
    const float* W2[3] = {(const float*)d_in[7],(const float*)d_in[11],(const float*)d_in[15]};
    const float* B2[3] = {(const float*)d_in[8],(const float*)d_in[12],(const float*)d_in[16]};
    const float* stem_w=(const float*)d_in[17]; const float* stem_b=(const float*)d_in[18];
    const float* cls1_w=(const float*)d_in[19]; const float* cls1_b=(const float*)d_in[20];
    const float* cls2_w=(const float*)d_in[21]; const float* cls2_b=(const float*)d_in[22];
    const float* reg1_w=(const float*)d_in[23]; const float* reg1_b=(const float*)d_in[24];
    const float* reg2_w=(const float*)d_in[25]; const float* reg2_b=(const float*)d_in[26];
    const float* clsp_w=(const float*)d_in[27]; const float* clsp_b=(const float*)d_in[28];
    const float* regp_w=(const float*)d_in[29]; const float* regp_b=(const float*)d_in[30];
    const float* objp_w=(const float*)d_in[31]; const float* objp_b=(const float*)d_in[32];

    char* ws = (char*)d_ws;  size_t off = 0;
    auto alloc = [&](size_t bytes)->char* {
        char* p = ws + off; off = (off + bytes + 255) & ~(size_t)255; return p; };

    const int din[3]  = {4, 64, 128};
    const int Kpad[3] = {64, 128, 192};     // concat width padded to 64
    const int Hl[3]   = {64, 128, 256};

    _Float16* wb1[3]; _Float16* wb2[3];
    for (int l = 0; l < 3; ++l) {
        wb1[l] = (_Float16*)alloc((size_t)Hl[l] * Kpad[l] * 2);
        wb2[l] = (_Float16*)alloc((size_t)Hl[l] * Hl[l] * 2);
    }
    _Float16* wstem = (_Float16*)alloc(256L * 256 * 2);
    _Float16* wc1   = (_Float16*)alloc(256L * 2304 * 2);
    _Float16* wc2   = (_Float16*)alloc(256L * 2304 * 2);
    _Float16* wr1   = (_Float16*)alloc(256L * 2304 * 2);
    _Float16* wr2   = (_Float16*)alloc(256L * 2304 * 2);
    _Float16* Amsg  = (_Float16*)alloc((size_t)N_MSG * 192 * 2);
    _Float16* G1    = (_Float16*)alloc((size_t)N_MSG * 256 * 2);
    _Float16* G2    = (_Float16*)alloc((size_t)N_MSG * 256 * 2);
    unsigned* keys  = (unsigned*)alloc((size_t)N_PTS * 256 * 4);
    _Float16* h16   = (_Float16*)alloc((size_t)N_PTS * 256 * 2);
    float*    fm32  = (float*)   alloc((size_t)NPIX * 256 * 4);
    _Float16* fm16  = (_Float16*)alloc((size_t)NPIX * 256 * 2);
    _Float16* S16   = (_Float16*)alloc((size_t)NPIX * 256 * 2);
    _Float16* C16a  = (_Float16*)alloc((size_t)NPIX * 256 * 2);
    _Float16* C16b  = (_Float16*)alloc((size_t)NPIX * 256 * 2);
    _Float16* R16a  = (_Float16*)alloc((size_t)NPIX * 256 * 2);
    _Float16* R16b  = (_Float16*)alloc((size_t)NPIX * 256 * 2);

    auto g1d = [](long n){ return (unsigned)((n + 255) / 256); };

    // ---- weight conversion ----
    for (int l = 0; l < 3; ++l) {
        cvt_lin_w<<<g1d((long)Hl[l]*Kpad[l]),256,0,stream>>>(W1[l], Hl[l], din[l]+3, Kpad[l], wb1[l]);
        cvt_lin_w<<<g1d((long)Hl[l]*Hl[l]),  256,0,stream>>>(W2[l], Hl[l], Hl[l],    Hl[l],   wb2[l]);
    }
    cvt_lin_w<<<g1d(256L*256),256,0,stream>>>(stem_w, 256, 256, 256, wstem);
    cvt_conv_w<<<g1d(256L*2304),256,0,stream>>>(cls1_w, wc1);
    cvt_conv_w<<<g1d(256L*2304),256,0,stream>>>(cls2_w, wc2);
    cvt_conv_w<<<g1d(256L*2304),256,0,stream>>>(reg1_w, wr1);
    cvt_conv_w<<<g1d(256L*2304),256,0,stream>>>(reg2_w, wr2);

    // ---- GNN backbone ----
    for (int l = 0; l < 3; ++l) {
        gather_msgs<<<g1d((long)N_MSG*Kpad[l]),256,0,stream>>>(
            (l == 0) ? x : nullptr, h16, din[l], Kpad[l], pos, ei, Amsg);
        dim3 grd1(N_MSG/64, Hl[l]/64);
        wmma_gemm<1><<<grd1,128,0,stream>>>(Amsg, N_MSG, Kpad[l], wb1[l], B1[l], G1, Hl[l]);
        wmma_gemm<0><<<grd1,128,0,stream>>>(G1,   N_MSG, Hl[l],   wb2[l], B2[l], G2, Hl[l]);
        long nh = (long)N_PTS * Hl[l];
        fill_u32<<<g1d(nh),256,0,stream>>>(keys, 0x007FFFFFu, nh);   // encode(-inf)
        segmax_scatter<<<N_MSG, Hl[l],0,stream>>>(G2, Hl[l], ei, keys);
        keys_to_f16<<<g1d(nh),256,0,stream>>>(keys, h16, nh);
    }

    // ---- points -> image ----
    fill_f32<<<g1d((long)NPIX*256),256,0,stream>>>(fm32, 0.f, (long)NPIX*256);
    scatter_image<<<N_PTS,256,0,stream>>>(h16, pos, batch, Kmat, fm32);
    f32_to_f16<<<g1d((long)NPIX*256),256,0,stream>>>(fm32, fm16, (long)NPIX*256);

    // ---- YOLOX head ----
    dim3 grdh(NPIX/64, 4);
    wmma_gemm<2><<<grdh,128,0,stream>>>(fm16, NPIX, 256, wstem, stem_b, S16, 256);
    wmma_conv3<2><<<grdh,128,0,stream>>>(S16,  wc1, cls1_b, C16a);
    wmma_conv3<2><<<grdh,128,0,stream>>>(C16a, wc2, cls2_b, C16b);
    wmma_conv3<2><<<grdh,128,0,stream>>>(S16,  wr1, reg1_b, R16a);
    wmma_conv3<2><<<grdh,128,0,stream>>>(R16a, wr2, reg2_b, R16b);

    pred_decode<<<g1d((long)NPIX*8),256,0,stream>>>(
        C16b, R16b, clsp_w, clsp_b, regp_w, regp_b, objp_w, objp_b, (float*)d_out);

    (void)in_sizes; (void)n_in; (void)out_size; (void)ws_size;
}
